// SpeciesGCN_6932077216413
// MI455X (gfx1250) — matrix operand order repaired and checked by reading, not compiled, not run on hardware
//
#include <hip/hip_runtime.h>
#include <hip/hip_bf16.h>
#include <math.h>

typedef __attribute__((ext_vector_type(16))) _Float16 v16h;
typedef __attribute__((ext_vector_type(8)))  float    v8f;
typedef __attribute__((ext_vector_type(4)))  float    v4f;

#define N_NODES 100000
#define DIM 64
#define MTILES (N_NODES / 16)   // 6250, exact

// ---------------------------------------------------------------------------
// init: zero agg[N*64], deg=1 (self-loop), agg2=0, and build W1^T in f16:
// w1h[n*64+k] = (f16) W1[k*64+n]  (so B-matrix columns are contiguous)
// ---------------------------------------------------------------------------
__global__ void gcn_init(float* __restrict__ agg, float* __restrict__ deg,
                         float* __restrict__ agg2, _Float16* __restrict__ w1h,
                         const float* __restrict__ W1) {
    long long i = (long long)blockIdx.x * blockDim.x + threadIdx.x;
    if (i < (long long)N_NODES * DIM) agg[i] = 0.0f;
    if (i < N_NODES) { deg[i] = 1.0f; agg2[i] = 0.0f; }
    if (i < DIM * DIM) {
        int n = (int)(i >> 6), k = (int)(i & 63);
        w1h[i] = (_Float16)W1[k * DIM + n];
    }
}

// deg[dst] += 1 per real edge (self-loops pre-counted by init)
__global__ void gcn_degree(const int* __restrict__ dst, int nEdges,
                           float* __restrict__ deg) {
    long long e = (long long)blockIdx.x * blockDim.x + threadIdx.x;
    if (e < nEdges) atomicAdd(&deg[dst[e]], 1.0f);
}

// deg -> deg^{-1/2} in place (deg >= 1 always due to self-loops)
__global__ void gcn_rsqrt(float* __restrict__ deg) {
    int i = blockIdx.x * blockDim.x + threadIdx.x;
    if (i < N_NODES) deg[i] = rsqrtf(deg[i]);
}

// ---------------------------------------------------------------------------
// hxw = x @ W1 via v_wmma_f32_16x16x32_f16.
// One wave per 16-row M-tile; 4 accumulators cover all 64 output columns.
// A (16x32 f16): lanes 0-15 row M=lane, K 0..15;
// lanes 16-31 same rows, K 16..31 (same contiguous order as B).
// B (32x16 f16): lane holds column n = lane%16; lanes<16 K=kc..kc+15,
// lanes>=16 K=kc+16..kc+31 (contiguous in transposed w1h).
// ---------------------------------------------------------------------------
__global__ __launch_bounds__(128) void gcn_gemm1_wmma(
        const float* __restrict__ x, const _Float16* __restrict__ w1h,
        float* __restrict__ hxw) {
    const int lane = threadIdx.x & 31;
    const int wave = threadIdx.x >> 5;
    const int tileM = blockIdx.x * 4 + wave;
    if (tileM >= MTILES) return;               // wave-uniform; EXEC stays all-1s

    const int r   = lane & 15;
    const int hi  = lane >> 4;                  // 0: low half, 1: high half
    const int row = tileM * 16 + r;
    const float* xrow = x + (long long)row * DIM;

    v8f acc0 = {}, acc1 = {}, acc2 = {}, acc3 = {};

    #pragma unroll
    for (int kc = 0; kc < DIM; kc += 32) {
        // ---- A chunk (16x32) ----
        const int kb0 = hi ? 16 : 0;
        v4f p0 = *(const v4f*)(xrow + kc + kb0);
        v4f p1 = *(const v4f*)(xrow + kc + kb0 + 4);
        v4f p2 = *(const v4f*)(xrow + kc + kb0 + 8);
        v4f p3 = *(const v4f*)(xrow + kc + kb0 + 12);
        v16h a;
        #pragma unroll
        for (int j = 0; j < 4; ++j) {
            a[j]      = (_Float16)p0[j];
            a[4 + j]  = (_Float16)p1[j];
            a[8 + j]  = (_Float16)p2[j];
            a[12 + j] = (_Float16)p3[j];
        }
        // ---- B chunks: column n = nt*16 + r, 16 contiguous f16 each ----
        const int koff = kc + (hi ? 16 : 0);
        v16h b0 = *(const v16h*)(w1h + (long long)(0 * 16 + r) * DIM + koff);
        v16h b1 = *(const v16h*)(w1h + (long long)(1 * 16 + r) * DIM + koff);
        v16h b2 = *(const v16h*)(w1h + (long long)(2 * 16 + r) * DIM + koff);
        v16h b3 = *(const v16h*)(w1h + (long long)(3 * 16 + r) * DIM + koff);

        acc0 = __builtin_amdgcn_wmma_f32_16x16x32_f16(false, a, false, b0,
                                                      (short)0, acc0, false, false);
        acc1 = __builtin_amdgcn_wmma_f32_16x16x32_f16(false, a, false, b1,
                                                      (short)0, acc1, false, false);
        acc2 = __builtin_amdgcn_wmma_f32_16x16x32_f16(false, a, false, b2,
                                                      (short)0, acc2, false, false);
        acc3 = __builtin_amdgcn_wmma_f32_16x16x32_f16(false, a, false, b3,
                                                      (short)0, acc3, false, false);
    }

    // D layout: VGPR rr -> M = rr + (hi?8:0), N = lane%16
    const int mbase = tileM * 16 + (hi ? 8 : 0);
    #pragma unroll
    for (int rr = 0; rr < 8; ++rr) {
        long long o = (long long)(mbase + rr) * DIM;
        hxw[o + 0 * 16 + r] = acc0[rr];
        hxw[o + 1 * 16 + r] = acc1[rr];
        hxw[o + 2 * 16 + r] = acc2[rr];
        hxw[o + 3 * 16 + r] = acc3[rr];
    }
}

// ---------------------------------------------------------------------------
// Layer-1 aggregation: agg[d] += hxw[s] * (dinv[s]*dinv[d]) per edge
// (incl. self-loops). 16 lanes per edge, float4 per lane -> 256B coalesced.
// ---------------------------------------------------------------------------
__global__ void gcn_scatter1(const int* __restrict__ src, const int* __restrict__ dst,
                             int nEdges, const float* __restrict__ dinv,
                             const float* __restrict__ hxw, float* __restrict__ agg) {
    long long t = (long long)blockIdx.x * blockDim.x + threadIdx.x;
    long long e = t >> 4;
    int j = (int)(t & 15);
    long long total = (long long)nEdges + N_NODES;
    if (e >= total) return;
    int s, d;
    if (e < nEdges) { s = src[e]; d = dst[e]; }
    else            { s = d = (int)(e - nEdges); }
    float w = dinv[s] * dinv[d];
    v4f v = *(const v4f*)(hxw + (long long)s * DIM + j * 4);
    float* ap = agg + (long long)d * DIM + j * 4;
    atomicAdd(ap + 0, v[0] * w);
    atomicAdd(ap + 1, v[1] * w);
    atomicAdd(ap + 2, v[2] * w);
    atomicAdd(ap + 3, v[3] * w);
}

// z[i] = sum_c relu(agg[i][c] + b1[c]) * W2[c]   (half-wave per node)
__global__ void gcn_relu_gemv(const float* __restrict__ agg, const float* __restrict__ b1,
                              const float* __restrict__ W2, float* __restrict__ z) {
    long long t = (long long)blockIdx.x * blockDim.x + threadIdx.x;
    long long node = t >> 4;
    int j = (int)(t & 15);
    if (node >= N_NODES) return;
    v4f a  = *(const v4f*)(agg + node * DIM + j * 4);
    v4f bb = *(const v4f*)(b1 + j * 4);
    v4f ww = *(const v4f*)(W2 + j * 4);
    float p = 0.0f;
    #pragma unroll
    for (int k = 0; k < 4; ++k) {
        float va = a[k] + bb[k];
        va = va > 0.0f ? va : 0.0f;
        p += va * ww[k];
    }
    p += __shfl_xor(p, 1);
    p += __shfl_xor(p, 2);
    p += __shfl_xor(p, 4);
    p += __shfl_xor(p, 8);
    if (j == 0) z[node] = p;
}

// Layer-2 aggregation: agg2[d] += z[s]*dinv[s]*dinv[d]
__global__ void gcn_scatter2(const int* __restrict__ src, const int* __restrict__ dst,
                             int nEdges, const float* __restrict__ dinv,
                             const float* __restrict__ z, float* __restrict__ agg2) {
    long long e = (long long)blockIdx.x * blockDim.x + threadIdx.x;
    long long total = (long long)nEdges + N_NODES;
    if (e >= total) return;
    int s, d;
    if (e < nEdges) { s = src[e]; d = dst[e]; }
    else            { s = d = (int)(e - nEdges); }
    atomicAdd(&agg2[d], z[s] * dinv[s] * dinv[d]);
}

__global__ void gcn_sigmoid(const float* __restrict__ agg2, const float* __restrict__ b2,
                            float* __restrict__ out) {
    int i = blockIdx.x * blockDim.x + threadIdx.x;
    if (i < N_NODES) {
        float v = agg2[i] + b2[0];
        out[i] = 1.0f / (1.0f + expf(-v));
    }
}

// ---------------------------------------------------------------------------
extern "C" void kernel_launch(void* const* d_in, const int* in_sizes, int n_in,
                              void* d_out, int out_size, void* d_ws, size_t ws_size,
                              hipStream_t stream) {
    const float* x  = (const float*)d_in[0];
    const int*   ei = (const int*)d_in[1];
    const float* W1 = (const float*)d_in[2];
    const float* b1 = (const float*)d_in[3];
    const float* W2 = (const float*)d_in[4];
    const float* b2 = (const float*)d_in[5];
    float* out = (float*)d_out;

    const int nEdges = in_sizes[1] / 2;
    const int* src = ei;
    const int* dst = ei + nEdges;

    // workspace layout (floats): hxw[N*64] | agg[N*64] | dinv[N] | agg2[N] | w1h(f16)
    float* hxw  = (float*)d_ws;
    float* agg  = hxw + (size_t)N_NODES * DIM;
    float* dinv = agg + (size_t)N_NODES * DIM;
    float* agg2 = dinv + N_NODES;
    _Float16* w1h = (_Float16*)(agg2 + N_NODES);   // byte offset 52,000,000 (32B aligned)
    float* z = hxw;   // reuse hxw's first N floats for layer-2 node scalars

    const int B = 256;
    long long nInit = (long long)N_NODES * DIM;
    gcn_init<<<(unsigned)((nInit + B - 1) / B), B, 0, stream>>>(agg, dinv, agg2, w1h, W1);

    gcn_degree<<<(nEdges + B - 1) / B, B, 0, stream>>>(dst, nEdges, dinv);
    gcn_rsqrt<<<(N_NODES + B - 1) / B, B, 0, stream>>>(dinv);

    gcn_gemm1_wmma<<<(MTILES + 3) / 4, 128, 0, stream>>>(x, w1h, hxw);

    long long sc1 = ((long long)nEdges + N_NODES) * 16;
    gcn_scatter1<<<(unsigned)((sc1 + B - 1) / B), B, 0, stream>>>(src, dst, nEdges, dinv, hxw, agg);

    long long gv = (long long)N_NODES * 16;
    gcn_relu_gemv<<<(unsigned)((gv + B - 1) / B), B, 0, stream>>>(agg, b1, W2, z);

    long long sc2 = (long long)nEdges + N_NODES;
    gcn_scatter2<<<(unsigned)((sc2 + B - 1) / B), B, 0, stream>>>(src, dst, nEdges, dinv, z, agg2);

    gcn_sigmoid<<<(N_NODES + B - 1) / B, B, 0, stream>>>(agg2, b2, out);
}
